// ModulatedDeformConv_69217692942560
// MI455X (gfx1250) — matrix operand order repaired and checked by reading, not compiled, not run on hardware
//
#include <hip/hip_runtime.h>
#include <hip/hip_bf16.h>

typedef __attribute__((ext_vector_type(16))) _Float16 v16h;
typedef __attribute__((ext_vector_type(8)))  _Float16 v8h;
typedef __attribute__((ext_vector_type(8)))  float    v8f;

#define B_    4
#define CIN   64
#define COUT  64
#define H_    128
#define W_    128
#define HW    (H_ * W_)          // 16384
#define KK    9
#define CKK   (CIN * KK)         // 576 (GEMM K dim, ordering k = kk*64 + cin)
#define NTILE 64                 // pixels per workgroup
#define COLSTR 584               // LDS row stride in halves: 1168 B (16B-aligned rows)
#define XT_OFFSET (128 * 1024)   // byte offset of transposed x inside d_ws

__device__ __forceinline__ v8h splat8(_Float16 s) {
    v8h v = {s, s, s, s, s, s, s, s};
    return v;
}

// ---------------------------------------------------------------------------
// Pack weight (COUT, CIN, 3, 3) f32 -> f16 wpk[cout][k], k = kk*64 + cin.
// ---------------------------------------------------------------------------
__global__ void pack_weights_f16(const float* __restrict__ w,
                                 _Float16* __restrict__ wpk) {
    int idx = blockIdx.x * 256 + threadIdx.x;
    if (idx < COUT * CKK) {
        int cout = idx / CKK;
        int k    = idx - cout * CKK;
        int kk   = k >> 6;
        int cin  = k & 63;
        wpk[idx] = (_Float16)w[(cout * CIN + cin) * KK + kk];
    }
}

// ---------------------------------------------------------------------------
// Transpose x (B, CIN, H, W) f32 -> xt (B, H, W, CIN) f16 (channel-contiguous).
// Reads coalesced along hw; writes 16B chunks per lane.
// ---------------------------------------------------------------------------
__global__ void transpose_x_f16(const float* __restrict__ x,
                                _Float16* __restrict__ xt) {
    const int gid = blockIdx.x * 256 + threadIdx.x;    // 65536 = B*HW
    const int b   = gid >> 14;
    const int hw  = gid & (HW - 1);
    const float* xs = x + (size_t)b * CIN * HW + hw;
    _Float16* dst = xt + (size_t)gid * CIN;
    #pragma unroll
    for (int ch = 0; ch < 8; ++ch) {
        v8h v;
        #pragma unroll
        for (int t = 0; t < 8; ++t)
            v[t] = (_Float16)xs[(size_t)(ch * 8 + t) * HW];
        *(v8h*)(dst + ch * 8) = v;
    }
}

// ---------------------------------------------------------------------------
// Fused deformable im2col (packed-f16 bilinear -> LDS) + WMMA GEMM (f32 accum).
// 256 threads = 8 wave32; 64 pixels x 64 couts of output per workgroup.
// Each wave: one cout tile x two pixel tiles (A fragment reused across 2 WMMAs).
// ---------------------------------------------------------------------------
__launch_bounds__(256)
__global__ void dcnv2_fused(const _Float16* __restrict__ xt,
                            const float* __restrict__ offset,
                            const float* __restrict__ mask,
                            const _Float16* __restrict__ wpk,
                            const float* __restrict__ bias,
                            float* __restrict__ out) {
    __shared__ __align__(32) _Float16 col[NTILE * COLSTR];   // ~73 KB

    const int tid   = threadIdx.x;
    const int pbase = blockIdx.x * NTILE;          // 64-aligned -> same batch b
    const int b     = pbase >> 14;

    // ---------------- Phase 1: branch-free deformable im2col into LDS ------
    for (int idx = tid; idx < NTILE * KK; idx += 256) {      // 576 tasks
        const int pixel = idx & (NTILE - 1);
        const int kk    = idx >> 6;                // 0..8
        const int p     = pbase + pixel;
        const int hw    = p & (HW - 1);
        const int h     = hw >> 7;
        const int wc    = hw & (W_ - 1);

        const float oi = offset[((size_t)b * (2 * KK) + 2 * kk) * HW + hw];
        const float oj = offset[((size_t)b * (2 * KK) + 2 * kk + 1) * HW + hw];
        const float mm = mask[((size_t)b * KK + kk) * HW + hw];

        const float ci = oi + (float)(h + (kk / 3) - 1);
        const float cj = oj + (float)(wc + (kk % 3) - 1);

        const float fi0 = floorf(ci), fj0 = floorf(cj);
        const int i0 = (int)fi0,       j0 = (int)fj0;
        const int i1 = (int)ceilf(ci), j1 = (int)ceilf(cj);
        const float fi = ci - fi0, fj = cj - fj0;
        const float gi = 1.0f - fi, gj = 1.0f - fj;

        const float vi0 = ((i0 >= 0) & (i0 < H_)) ? 1.0f : 0.0f;
        const float vi1 = ((i1 >= 0) & (i1 < H_)) ? 1.0f : 0.0f;
        const float vj0 = ((j0 >= 0) & (j0 < W_)) ? 1.0f : 0.0f;
        const float vj1 = ((j1 >= 0) & (j1 < W_)) ? 1.0f : 0.0f;

        // corner weights with validity and modulation mask folded in
        const v8h w00 = splat8((_Float16)(gi * gj * mm * vi0 * vj0));
        const v8h w01 = splat8((_Float16)(gi * fj * mm * vi0 * vj1));
        const v8h w10 = splat8((_Float16)(fi * gj * mm * vi1 * vj0));
        const v8h w11 = splat8((_Float16)(fi * fj * mm * vi1 * vj1));

        const int i0c = min(max(i0, 0), H_ - 1), j0c = min(max(j0, 0), W_ - 1);
        const int i1c = min(max(i1, 0), H_ - 1), j1c = min(max(j1, 0), W_ - 1);

        const _Float16* xb  = xt + (size_t)b * HW * CIN;
        const _Float16* c00 = xb + (size_t)(i0c * W_ + j0c) * CIN;  // 128B aligned
        const _Float16* c01 = xb + (size_t)(i0c * W_ + j1c) * CIN;
        const _Float16* c10 = xb + (size_t)(i1c * W_ + j0c) * CIN;
        const _Float16* c11 = xb + (size_t)(i1c * W_ + j1c) * CIN;

        _Float16* dst = &col[pixel * COLSTR + kk * CIN];
        #pragma unroll
        for (int t = 0; t < 8; ++t) {              // 8 x (4 b128 loads + pk_fma chain)
            const v8h p00 = *(const v8h*)(c00 + t * 8);
            const v8h p01 = *(const v8h*)(c01 + t * 8);
            const v8h p10 = *(const v8h*)(c10 + t * 8);
            const v8h p11 = *(const v8h*)(c11 + t * 8);
            v8h r = p00 * w00;
            r = r + p01 * w01;
            r = r + p10 * w10;
            r = r + p11 * w11;
            *(v8h*)(dst + t * 8) = r;
        }
    }
    __syncthreads();

    // ---------------- Phase 2: WMMA GEMM, two 16x16 tiles per wave ----------
    const int wave = tid >> 5;
    const int lane = tid & 31;
    const int ct   = wave & 3;                     // cout tile 0..3
    const int pt0  = (wave >> 2) * 2;              // pixel tiles {pt0, pt0+1}
    const int mrow = lane & 15;
    const bool hiL = lane >= 16;

    const _Float16* wrow  = wpk + (size_t)(ct * 16 + mrow) * CKK;
    const _Float16* crow0 = &col[(pt0 * 16 + mrow) * COLSTR];
    const _Float16* crow1 = crow0 + 16 * COLSTR;

    v8f acc0 = {}, acc1 = {};
    #pragma unroll
    for (int ks = 0; ks < CKK / 32; ++ks) {        // 18 K-steps, 36 WMMAs
        const int kbase = ks * 32;
        // A: 16-bit 16x32 layout (lanes 0-15: K{0..7,16..23}; 16-31: K{8..15,24..31})
        const int kA0 = kbase + (hiL ? 8  : 0);
        const int kA1 = kbase + (hiL ? 24 : 16);
        const v8h a0 = *(const v8h*)(wrow + kA0);
        const v8h a1 = *(const v8h*)(wrow + kA1);
        v16h A;
        #pragma unroll
        for (int t = 0; t < 8; ++t) { A[t] = a0[t]; A[8 + t] = a1[t]; }

        // B: 16-bit 32x16 layout (lanes 0-15: K{0..7,16..23}; 16-31: K{8..15,24..31})
        const int kB0 = kbase + (hiL ? 8 : 0);
        const v8h b00 = *(const v8h*)(crow0 + kB0);
        const v8h b01 = *(const v8h*)(crow0 + kB0 + 16);
        const v8h b10 = *(const v8h*)(crow1 + kB0);
        const v8h b11 = *(const v8h*)(crow1 + kB0 + 16);
        v16h B0, B1;
        #pragma unroll
        for (int t = 0; t < 8; ++t) {
            B0[t] = b00[t]; B0[8 + t] = b01[t];
            B1[t] = b10[t]; B1[8 + t] = b11[t];
        }

        acc0 = __builtin_amdgcn_wmma_f32_16x16x32_f16(
            false, A, false, B0, (short)0, acc0, false, false);
        acc1 = __builtin_amdgcn_wmma_f32_16x16x32_f16(
            false, A, false, B1, (short)0, acc1, false, false);
    }

    // ---------------- Store: lane%16 = N(pixel), VGPR r -> M(cout) ----------
    #pragma unroll
    for (int tile = 0; tile < 2; ++tile) {
        const v8f acc = tile ? acc1 : acc0;
        const int pglob = pbase + (pt0 + tile) * 16 + mrow;
        const int ob    = pglob >> 14;
        const int ohw   = pglob & (HW - 1);
        #pragma unroll
        for (int r = 0; r < 8; ++r) {
            const int cout = ct * 16 + r + (hiL ? 8 : 0);
            out[((size_t)ob * COUT + cout) * HW + ohw] = acc[r] + bias[cout];
        }
    }
}

// ---------------------------------------------------------------------------
extern "C" void kernel_launch(void* const* d_in, const int* in_sizes, int n_in,
                              void* d_out, int out_size, void* d_ws, size_t ws_size,
                              hipStream_t stream) {
    const float* x      = (const float*)d_in[0];
    const float* offset = (const float*)d_in[1];
    const float* mask   = (const float*)d_in[2];
    const float* weight = (const float*)d_in[3];
    const float* bias   = (const float*)d_in[4];
    float*       out    = (float*)d_out;

    _Float16* wpk = (_Float16*)d_ws;                                // 72 KB
    _Float16* xt  = (_Float16*)((char*)d_ws + XT_OFFSET);           // 8 MB

    pack_weights_f16<<<(COUT * CKK + 255) / 256, 256, 0, stream>>>(weight, wpk);
    transpose_x_f16<<<(B_ * HW) / 256, 256, 0, stream>>>(x, xt);

    const int nblocks = (B_ * HW) / NTILE;                          // 1024
    dcnv2_fused<<<nblocks, 256, 0, stream>>>(xt, offset, mask, wpk, bias, out);
}
